// SymbolicGNN_63024350101869
// MI455X (gfx1250) — matrix operand order repaired and checked by reading, not compiled, not run on hardware
//
#include <hip/hip_runtime.h>
#include <hip/hip_bf16.h>

// ---------------------------------------------------------------------------
// SymbolicGNN on MI455X (gfx1250): edge-MLP as wave32 WMMA bf16 GEMMs.
//   h (N x 64, f32)  double-buffered in d_ws
//   per 16-edge tile: X (16x192 bf16) @ W1 (192x128) -> GELU -> @ W2 (128x64)
//   scatter-add msg into h via global_atomic_add_f32
// Weights staged into LDS via GLOBAL_LOAD_ASYNC_TO_LDS_B128 (ASYNCcnt path).
// ---------------------------------------------------------------------------

typedef __attribute__((ext_vector_type(16))) __bf16      v16bf;
typedef __attribute__((ext_vector_type(8)))  float       v8f;
typedef __attribute__((ext_vector_type(8)))  unsigned int v8u;

#define N_NODES 100000
#define ES 64
#define KTOT 192     // 3*ES
#define HID 128      // 2*ES
#define DL 256

// f32 -> bf16 round-to-nearest-even
__device__ __forceinline__ unsigned short f2bf(float f) {
    unsigned u = __builtin_bit_cast(unsigned, f);
    unsigned r = u + 0x7FFFu + ((u >> 16) & 1u);
    return (unsigned short)(r >> 16);
}
__device__ __forceinline__ unsigned int packbf(float a, float b) {
    return (unsigned)f2bf(a) | ((unsigned)f2bf(b) << 16);
}
// Build a 32-byte WMMA fragment from two 16B-aligned LDS chunks
__device__ __forceinline__ v16bf load_frag(const uint4* p0, const uint4* p1) {
    uint4 a = *p0, b = *p1;
    v8u u;
    u[0]=a.x; u[1]=a.y; u[2]=a.z; u[3]=a.w;
    u[4]=b.x; u[5]=b.y; u[6]=b.z; u[7]=b.w;
    return __builtin_bit_cast(v16bf, u);
}
// Generic LDS pointer -> LDS byte offset (flat LDS aperture keeps the offset
// in the low 32 bits; hardware truncates, see ISA 10.2 aperture rules).
__device__ __forceinline__ unsigned lds_off(const void* p) {
    return (unsigned)(unsigned long long)p;
}
// Async DMA: 16 bytes global -> LDS, per lane, tracked by ASYNCcnt.
__device__ __forceinline__ void async_g2l_b128(unsigned lds_byte_off, const void* gaddr) {
    asm volatile("global_load_async_to_lds_b128 %0, %1, off"
                 :: "v"(lds_byte_off), "v"((unsigned long long)gaddr)
                 : "memory");
}

// ---------------------------------------------------------------------------
// Pack msg_w1 (L,192,128) & msg_w2 (L,128,64) f32 -> bf16 in B-operand layout:
// per (kb,nb) 32x16 tile: lane = khalf*16 + n holds K pairs (2j,2j+1) per dword.
// Also zeroes the column-sum accumulator.
// ---------------------------------------------------------------------------
__global__ void pack_weights_kernel(const float* __restrict__ w1,
                                    const float* __restrict__ w2,
                                    unsigned short* __restrict__ w1p,
                                    unsigned short* __restrict__ w2p,
                                    float* __restrict__ colsum) {
    int tid = blockIdx.x * blockDim.x + threadIdx.x;
    if (tid < ES) colsum[tid] = 0.f;
    if (tid < 2 * KTOT * HID) {
        int l = tid / (KTOT * HID), rem = tid % (KTOT * HID);
        int k = rem / HID, n = rem % HID;
        int kb = k >> 5, kk = k & 31, kh = (kk >> 3) & 1, kp = (kk & 7) | ((kk >> 4) << 3);
        int j = kp >> 1, pos = kp & 1;
        int nb = n >> 4, nn = n & 15, lanep = kh * 16 + nn;
        size_t dst = ((((size_t)l * 6 + kb) * 8 + nb) * 32 + lanep) * 16 + j * 2 + pos;
        w1p[dst] = f2bf(w1[tid]);
    }
    if (tid < 2 * HID * ES) {
        int l = tid / (HID * ES), rem = tid % (HID * ES);
        int k = rem / ES, n = rem % ES;
        int kb = k >> 5, kk = k & 31, kh = (kk >> 3) & 1, kp = (kk & 7) | ((kk >> 4) << 3);
        int j = kp >> 1, pos = kp & 1;
        int nb = n >> 4, nn = n & 15, lanep = kh * 16 + nn;
        size_t dst = ((((size_t)l * 4 + kb) * 4 + nb) * 32 + lanep) * 16 + j * 2 + pos;
        w2p[dst] = f2bf(w2[tid]);
    }
}

// h0 = hA = hB = sym_emb[0:N] (node id < vocab, so id % V == id)
__global__ void init_h_kernel(const float* __restrict__ sym,
                              float* __restrict__ hA, float* __restrict__ hB, int n) {
    for (int i = blockIdx.x * blockDim.x + threadIdx.x; i < n; i += gridDim.x * blockDim.x) {
        float v = sym[i];
        hA[i] = v; hB[i] = v;
    }
}

// ---------------------------------------------------------------------------
// Main edge kernel: 8 waves/block, each wave processes 4 tiles of 16 edges.
// LDS: [W1 48K][W2 16K][b1 512][b2 256][pad][X 6K x8][HM 4K x8] = 148480 B
// ---------------------------------------------------------------------------
#define LDS_W2   49152
#define LDS_B1   65536
#define LDS_B2   66048
#define LDS_X    66560
#define LDS_HM   115712
#define LDS_TOT  148480
#define TILES_PER_WAVE 4

__global__ __launch_bounds__(256)
void edge_layer_kernel(const int* __restrict__ subj, const int* __restrict__ pred,
                       const int* __restrict__ obj,
                       const float* __restrict__ h_read, float* __restrict__ h_acc,
                       const float* __restrict__ edge_emb,
                       const unsigned short* __restrict__ w1p,
                       const unsigned short* __restrict__ w2p,
                       const float* __restrict__ b1, const float* __restrict__ b2,
                       int n_tiles) {
    extern __shared__ char smem[];
    unsigned int* w1l = (unsigned int*)smem;               // 12288 dwords
    unsigned int* w2l = (unsigned int*)(smem + LDS_W2);    // 4096 dwords
    float* b1l = (float*)(smem + LDS_B1);
    float* b2l = (float*)(smem + LDS_B2);

    const int tid = threadIdx.x, lane = tid & 31, wave = tid >> 5;

    // ---- stage weights into LDS with async DMA (no VGPR round-trip) ----
    {
        const unsigned base_w1 = lds_off(smem);
        const char* g1 = (const char*)w1p;
        #pragma unroll
        for (int i = 0; i < 12; ++i) {              // 3072 x 16B = 48KB, uniform
            int c = tid + i * 256;
            async_g2l_b128(base_w1 + c * 16, g1 + (size_t)c * 16);
        }
        const unsigned base_w2 = lds_off(smem + LDS_W2);
        const char* g2 = (const char*)w2p;
        #pragma unroll
        for (int i = 0; i < 4; ++i) {               // 1024 x 16B = 16KB, uniform
            int c = tid + i * 256;
            async_g2l_b128(base_w2 + c * 16, g2 + (size_t)c * 16);
        }
        if (tid < HID) b1l[tid] = b1[tid];
        if (tid < ES)  b2l[tid] = b2[tid];
        asm volatile("s_wait_asynccnt 0" ::: "memory");
    }
    __syncthreads();

    unsigned short* X  = (unsigned short*)(smem + LDS_X  + wave * 6144);  // 16 x 192 bf16
    unsigned short* HM = (unsigned short*)(smem + LDS_HM + wave * 4096);  // 16 x 128 bf16

    const int row = lane & 15;        // M row this lane owns for the A layout
    const int sel = lane >> 4;        // K-half select (A) / M-half select (C)
    const int nn  = lane & 15;        // N column within a 16-wide tile

    for (int t = 0; t < TILES_PER_WAVE; ++t) {
        const int tile = (blockIdx.x * 8 + wave) * TILES_PER_WAVE + t;
        if (tile >= n_tiles) break;   // uniform per wave -> EXEC stays full
        const int base16 = tile * 16;

        // ---- gather: lanes (2r, 2r+1) build row r of X = [h[subj] | e | h[obj]] ----
        {
            const int r = lane >> 1, half = lane & 1;
            const int eidx = base16 + r;
            const int pr = pred[eidx];
            unsigned int* xr = (unsigned int*)(X + r * KTOT);
            const float4* ep = (const float4*)(edge_emb + (size_t)pr * ES);
            if (half == 0) {
                const int sj = subj[eidx];
                const float4* hs = (const float4*)(h_read + (size_t)sj * ES);
                #pragma unroll
                for (int q = 0; q < 16; ++q) {
                    float4 f = hs[q];
                    xr[2*q]   = packbf(f.x, f.y);
                    xr[2*q+1] = packbf(f.z, f.w);
                }
                #pragma unroll
                for (int q = 0; q < 8; ++q) {
                    float4 f = ep[q];
                    xr[32 + 2*q] = packbf(f.x, f.y);
                    xr[33 + 2*q] = packbf(f.z, f.w);
                }
            } else {
                const int ob = obj[eidx];
                const float4* ho = (const float4*)(h_read + (size_t)ob * ES);
                #pragma unroll
                for (int q = 0; q < 8; ++q) {
                    float4 f = ep[8 + q];
                    xr[48 + 2*q] = packbf(f.x, f.y);
                    xr[49 + 2*q] = packbf(f.z, f.w);
                }
                #pragma unroll
                for (int q = 0; q < 16; ++q) {
                    float4 f = ho[q];
                    xr[64 + 2*q] = packbf(f.x, f.y);
                    xr[65 + 2*q] = packbf(f.z, f.w);
                }
            }
            // prefetch next tile's gather rows so HBM/L2 fetch overlaps the WMMAs
            if (t + 1 < TILES_PER_WAVE && tile + 1 < n_tiles) {
                const int e2 = base16 + 16 + r;
                if (half == 0) {
                    __builtin_prefetch(h_read + (size_t)subj[e2] * ES, 0, 3);
                } else {
                    __builtin_prefetch(h_read + (size_t)obj[e2] * ES, 0, 3);
                    __builtin_prefetch(edge_emb + (size_t)pred[e2] * ES, 0, 3);
                }
            }
        }
        // in-wave LDS ordering: gather stores before A-fragment loads
        asm volatile("s_wait_dscnt 0" ::: "memory");

        // ---- GEMM1: (16x192) @ (192x128) -> c[8] f32 accumulators ----
        v8f c[8];
        #pragma unroll
        for (int nb = 0; nb < 8; ++nb) c[nb] = (v8f){0,0,0,0,0,0,0,0};

        const unsigned short* arow = X + row * KTOT + sel * 8;
        #pragma unroll
        for (int kb = 0; kb < 6; ++kb) {
            v16bf a = load_frag((const uint4*)(arow + kb * 32),
                                (const uint4*)(arow + kb * 32 + 16));
            const unsigned int* wb = w1l + kb * 8 * 256 + lane * 8;
            #pragma unroll
            for (int nb = 0; nb < 8; ++nb) {
                v16bf b = load_frag((const uint4*)(wb + nb * 256),
                                    (const uint4*)(wb + nb * 256 + 4));
                c[nb] = __builtin_amdgcn_wmma_f32_16x16x32_bf16(
                            false, a, false, b, (short)0, c[nb], false, false);
            }
        }

        // ---- bias + exact-erf GELU, re-layout into HM (16x128 bf16, row major) ----
        {
            const int mb = sel * 8;
            #pragma unroll
            for (int nb = 0; nb < 8; ++nb) {
                float bias = b1l[nb * 16 + nn];
                #pragma unroll
                for (int i = 0; i < 8; ++i) {
                    float x = c[nb][i] + bias;
                    x = 0.5f * x * (1.0f + erff(x * 0.70710678118654752f));
                    HM[(mb + i) * HID + nb * 16 + nn] = f2bf(x);
                }
            }
        }
        asm volatile("s_wait_dscnt 0" ::: "memory");

        // ---- GEMM2: (16x128) @ (128x64) -> d[4] ----
        v8f d[4];
        #pragma unroll
        for (int nb = 0; nb < 4; ++nb) d[nb] = (v8f){0,0,0,0,0,0,0,0};

        const unsigned short* arow2 = HM + row * HID + sel * 8;
        #pragma unroll
        for (int kb = 0; kb < 4; ++kb) {
            v16bf a = load_frag((const uint4*)(arow2 + kb * 32),
                                (const uint4*)(arow2 + kb * 32 + 16));
            const unsigned int* wb = w2l + kb * 4 * 256 + lane * 8;
            #pragma unroll
            for (int nb = 0; nb < 4; ++nb) {
                v16bf b = load_frag((const uint4*)(wb + nb * 256),
                                    (const uint4*)(wb + nb * 256 + 4));
                d[nb] = __builtin_amdgcn_wmma_f32_16x16x32_bf16(
                            false, a, false, b, (short)0, d[nb], false, false);
            }
        }

        // ---- bias + scatter-add msg into h_acc[obj] ----
        {
            int oidx[8];
            #pragma unroll
            for (int i = 0; i < 8; ++i) oidx[i] = obj[base16 + sel * 8 + i];
            #pragma unroll
            for (int nb = 0; nb < 4; ++nb) {
                const int n = nb * 16 + nn;
                const float bias = b2l[n];
                #pragma unroll
                for (int i = 0; i < 8; ++i) {
                    atomicAdd(h_acc + (size_t)oidx[i] * ES + n, d[nb][i] + bias);
                }
            }
        }
    }
}

// ---------------------------------------------------------------------------
// Column mean + final projection
// ---------------------------------------------------------------------------
__global__ void col_reduce_kernel(const float* __restrict__ h,
                                  float* __restrict__ colsum, int n_nodes) {
    __shared__ float red[256];
    const int col = threadIdx.x & 63, grp = threadIdx.x >> 6;
    float acc = 0.f;
    for (int node = blockIdx.x * 4 + grp; node < n_nodes; node += gridDim.x * 4)
        acc += h[(size_t)node * ES + col];
    red[threadIdx.x] = acc;
    __syncthreads();
    if (threadIdx.x < 64) {
        float s = red[threadIdx.x] + red[threadIdx.x + 64] +
                  red[threadIdx.x + 128] + red[threadIdx.x + 192];
        atomicAdd(&colsum[col], s);
    }
}

__global__ void final_proj_kernel(const float* __restrict__ colsum,
                                  const float* __restrict__ W,
                                  const float* __restrict__ b,
                                  float* __restrict__ out, float invN) {
    const int j = threadIdx.x;  // 256
    float acc = b[j];
    #pragma unroll
    for (int k = 0; k < ES; ++k) acc += (colsum[k] * invN) * W[k * DL + j];
    out[j] = acc;
}

// ---------------------------------------------------------------------------
extern "C" void kernel_launch(void* const* d_in, const int* in_sizes, int n_in,
                              void* d_out, int out_size, void* d_ws, size_t ws_size,
                              hipStream_t stream) {
    const int*   subj = (const int*)d_in[0];
    const int*   pred = (const int*)d_in[1];
    const int*   obj  = (const int*)d_in[2];
    const float* sym  = (const float*)d_in[3];
    const float* eemb = (const float*)d_in[4];
    const float* w1   = (const float*)d_in[5];
    const float* b1   = (const float*)d_in[6];
    const float* w2   = (const float*)d_in[7];
    const float* b2   = (const float*)d_in[8];
    const float* tw   = (const float*)d_in[9];
    const float* tb   = (const float*)d_in[10];

    const int E = in_sizes[0];
    const int N = N_NODES;
    const size_t hbytes = (size_t)N * ES * sizeof(float);   // 25,600,000 B

    char* ws = (char*)d_ws;
    float*          hA   = (float*)(ws);
    float*          hB   = (float*)(ws + hbytes);
    unsigned short* w1p  = (unsigned short*)(ws + 2 * hbytes);                 // 2*24576 bf16
    unsigned short* w2p  = (unsigned short*)(ws + 2 * hbytes + 98304);         // 2*8192 bf16
    float*          csum = (float*)(ws + 2 * hbytes + 98304 + 32768);

    // pack weights (49152 threads covers both W1 and W2) + zero colsum
    pack_weights_kernel<<<192, 256, 0, stream>>>(w1, w2, w1p, w2p, csum);
    init_h_kernel<<<1024, 256, 0, stream>>>(sym, hA, hB, N * ES);

    const int n_tiles = (E + 15) / 16;                       // 62500
    const int blocks  = (n_tiles + 8 * TILES_PER_WAVE - 1) / (8 * TILES_PER_WAVE);

    // layer 0: read hA, accumulate into hB (hB pre-initialized to h0)
    edge_layer_kernel<<<blocks, 256, LDS_TOT, stream>>>(
        subj, pred, obj, hA, hB, eemb, w1p, w2p, b1, b2, n_tiles);
    // hA := h1
    hipMemcpyAsync(hA, hB, hbytes, hipMemcpyDeviceToDevice, stream);
    // layer 1: read hB (=h1), accumulate into hA
    edge_layer_kernel<<<blocks, 256, LDS_TOT, stream>>>(
        subj, pred, obj, hB, hA, eemb, w1p + 24576, w2p + 8192,
        b1 + HID, b2 + ES, n_tiles);

    col_reduce_kernel<<<256, 256, 0, stream>>>(hA, csum, N);
    final_proj_kernel<<<1, 256, 0, stream>>>(csum, tw, tb, (float*)d_out, 1.0f / (float)N);
}
